// MultiheadAttention_38225208934675
// MI455X (gfx1250) — matrix operand order repaired and checked by reading, not compiled, not run on hardware
//
#include <hip/hip_runtime.h>

#define NN   50000
#define EE   800000
#define DIN  128
#define SHD  16
#define EAD  16
#define NH   8
#define DQv  64
#define DKv  64
#define DVv  128
#define HIDv 64

typedef __attribute__((ext_vector_type(16))) __bf16         v16bf;
typedef __attribute__((ext_vector_type(16))) unsigned short v16u;
typedef __attribute__((ext_vector_type(8)))  float          v8f;

// ---- helpers ---------------------------------------------------------------

__device__ __forceinline__ unsigned short f2bf(float f) {
  return __builtin_bit_cast(unsigned short, (__bf16)f);   // hw cvt, RNE
}

__device__ __forceinline__ v8f wmma_bf16(v16u a, v16u b, v8f c) {
  return __builtin_amdgcn_wmma_f32_16x16x32_bf16(
      false, __builtin_bit_cast(v16bf, a),
      false, __builtin_bit_cast(v16bf, b),
      (short)0, c, false, false);
}

// A tile 16x32 (row-major bf16-bits in LDS, row stride lda). ISA 16-bit A layout:
// lanes 0-15 / 16-31 = K-halves; VGPR j<4 -> K=2j, j>=4 -> K=16+2(j-4).
// Lowers to 2x ds_load_b128 per operand.
__device__ __forceinline__ v16u ldA(const unsigned short* A, int lda, int lane) {
  int m = lane & 15, h = (lane >> 4) & 1;
  const unsigned short* row = A + m * lda + 8 * h;
  v16u a;
#pragma unroll
  for (int j = 0; j < 8; ++j) {
    int k = ((j & 4) << 2) + 2 * (j & 3);
    a[2 * j]     = row[k];
    a[2 * j + 1] = row[k + 1];
  }
  return a;
}

// B operand from a pre-packed buffer: [(kt*ntiles+nt)*32 + lane]*16 contiguous
// ushorts already in the WMMA per-lane layout -> 2x global_load_b128.
__device__ __forceinline__ v16u ldBp(const unsigned short* Bp, int tile, int lane) {
  return *(const v16u*)(Bp + ((size_t)tile * 32 + lane) * 16);
}

__device__ __forceinline__ void atomicMaxF(float* addr, float v) {
  if (v >= 0.f) atomicMax((int*)addr, __float_as_int(v));
  else          atomicMin((unsigned int*)addr, __float_as_uint(v));
}

// Pack a row-major [K][N] f32 weight into per-lane WMMA B layout (bf16 bits).
// tstride != 0 means the logical B[k][n] = W[n*tstride + k] (transposed source).
// Rows k >= K are zero (handles K=16 padded to 32).
__device__ void packB(const float* W, int K, int N, int ktiles, int tstride,
                      unsigned short* out, int tid) {
  int ntiles = N >> 4;
  int total = ktiles * ntiles * 32 * 16;
  for (int idx = tid; idx < total; idx += 256) {
    int el   = idx & 15;
    int lane = (idx >> 4) & 31;
    int tile = idx >> 9;
    int nt = tile % ntiles, kt = tile / ntiles;
    int j = el >> 1, odd = el & 1;
    int n = lane & 15, h = (lane >> 4) & 1;
    int k = kt * 32 + 8 * h + (el & 7) + ((el >> 3) << 4); (void)j; (void)odd;
    float v = 0.f;
    if (k < K) v = tstride ? W[(nt * 16 + n) * tstride + k] : W[k * N + nt * 16 + n];
    out[idx] = f2bf(v);
  }
}

// ---- weight prep: fold Wdot into Wq, pack everything -----------------------

__global__ void prep_kernel(const float* Wq, const float* Wdot,
                            const float* Wang_k, const float* Wlin_k,
                            const float* W1k, const float* W2k,
                            const float* Wang_v, const float* Wlin_v,
                            const float* W1v, const float* W2v, const float* Wout,
                            float* Wq2f,
                            unsigned short* Wq2p, unsigned short* WangkTp,
                            unsigned short* Wlinkp, unsigned short* W1kp,
                            unsigned short* W2kp, unsigned short* WangvTp,
                            unsigned short* Wlinvp, unsigned short* W1vp,
                            unsigned short* W2vp, unsigned short* Woutp) {
  int t = threadIdx.x;
  for (int idx = t; idx < DIN * DQv; idx += 256) {      // Wq2 = Wq @ blockdiag(Wdot)
    int i = idx >> 6, col = idx & 63, h = col >> 3, kk = col & 7;
    float s = 0.f;
    for (int q = 0; q < 8; ++q) s += Wq[i * DQv + h * 8 + q] * Wdot[q * 8 + kk];
    Wq2f[idx] = s;
  }
  __syncthreads();
  packB(Wq2f,   DIN,  DQv, 4, 0,   Wq2p,    t);
  packB(Wang_k, SHD,  DIN, 1, SHD, WangkTp, t);   // (edge_sh @ Wang_k^T)
  packB(Wlin_k, DIN,  DKv, 4, 0,   Wlinkp,  t);
  packB(W1k,    EAD,  HIDv,1, 0,   W1kp,    t);
  packB(W2k,    HIDv, DKv, 2, 0,   W2kp,    t);
  packB(Wang_v, SHD,  DIN, 1, SHD, WangvTp, t);
  packB(Wlin_v, DIN,  DVv, 4, 0,   Wlinvp,  t);
  packB(W1v,    EAD,  HIDv,1, 0,   W1vp,    t);
  packB(W2v,    HIDv, DVv, 2, 0,   W2vp,    t);
  packB(Wout,   DIN,  DVv, 4, 0,   Woutp,   t);
}

__global__ void init_kernel(float* amax, float* denom, float* acc) {
  long long idx = (long long)blockIdx.x * blockDim.x + threadIdx.x;
  if (idx < (long long)NN * DVv) acc[idx] = 0.f;
  if (idx < (long long)NN * NH) { amax[idx] = -3.0e38f; denom[idx] = 0.f; }
}

// ---- qW = node_attr @ Wq2 (WMMA, 32 rows / block, 8 waves) -----------------

__global__ __launch_bounds__(256) void qw_kernel(const float* node_attr,
                                                 const unsigned short* Wq2p, float* qW) {
  __shared__ unsigned short As[32][DIN];
  int m0 = blockIdx.x * 32, tid = threadIdx.x;
  const bool full = (m0 + 32 <= NN);        // block-uniform: only last block is partial
  if (full) {
    for (int idx = tid; idx < 32 * DIN; idx += 256) {
      int r = idx >> 7, c = idx & 127;
      As[r][c] = f2bf(node_attr[(long long)(m0 + r) * DIN + c]);
    }
  } else {
    for (int idx = tid; idx < 32 * DIN; idx += 256) {
      int r = idx >> 7, c = idx & 127, gm = m0 + r;
      As[r][c] = (gm < NN) ? f2bf(node_attr[(long long)gm * DIN + c]) : (unsigned short)0;
    }
  }
  __syncthreads();
  int wave = tid >> 5, lane = tid & 31, mt = wave >> 2, nt = wave & 3;
  v8f c = {};
#pragma unroll
  for (int kk = 0; kk < 4; ++kk) {
    v16u a = ldA(&As[mt * 16][kk * 32], DIN, lane);
    v16u b = ldBp(Wq2p, kk * 4 + nt, lane);
    c = wmma_bf16(a, b, c);
  }
  int n = lane & 15, hh = (lane >> 4) & 1;
  int gm0 = m0 + mt * 16 + 8 * hh;
  if (full) {
#pragma unroll
    for (int r = 0; r < 8; ++r)
      qW[(long long)(gm0 + r) * DQv + nt * 16 + n] = c[r];
  } else {
#pragma unroll
    for (int r = 0; r < 8; ++r)
      if (gm0 + r < NN) qW[(long long)(gm0 + r) * DQv + nt * 16 + n] = c[r];
  }
}

// ---- pass A: per-edge key pipeline + alpha + atomic max --------------------

__global__ __launch_bounds__(128) void passA_kernel(
    const float* node_attr, const float* edge_attr, const float* edge_sh,
    const float* b1k, const int* edge_index,
    const unsigned short* WangkTp, const unsigned short* W1kp,
    const unsigned short* W2kp, const unsigned short* Wlinkp,
    const float* qW, float* alpha, float* amax) {
  __shared__ float          Xs[16][DIN];
  __shared__ unsigned short Ax[16][DIN];
  __shared__ unsigned short Ash[16][32];
  __shared__ unsigned short Aea[16][32];
  __shared__ unsigned short Hid[16][HIDv];
  __shared__ float          Rad[16][DKv];
  __shared__ float          Kt[16][DKv];
  __shared__ int Src[16], Dst[16];

  const int e0 = blockIdx.x * 16, tid = threadIdx.x;
  if (tid < 16) { Src[tid] = edge_index[e0 + tid]; Dst[tid] = edge_index[EE + e0 + tid]; }
  for (int idx = tid; idx < 16 * 32; idx += 128) {
    int r = idx >> 5, c = idx & 31;
    Aea[r][c] = (c < EAD) ? f2bf(edge_attr[(long long)(e0 + r) * EAD + c]) : (unsigned short)0;
    Ash[r][c] = (c < SHD) ? f2bf(edge_sh[(long long)(e0 + r) * SHD + c])  : (unsigned short)0;
  }
  __syncthreads();
  for (int idx = tid; idx < 16 * DIN; idx += 128) {     // gather x_src (L2-resident)
    int r = idx >> 7, c = idx & 127;
    Xs[r][c] = node_attr[(long long)Src[r] * DIN + c];
  }
  __syncthreads();

  const int wave = tid >> 5, lane = tid & 31;
  const int n = lane & 15, hh = (lane >> 4) & 1;

  for (int t = wave; t < 8; t += 4) {                   // g = sh @ WangT; Ax = bf16(x*g)
    v16u a = ldA(&Ash[0][0], 32, lane);
    v16u b = ldBp(WangkTp, t, lane);
    v8f c = {};
    c = wmma_bf16(a, b, c);
#pragma unroll
    for (int r = 0; r < 8; ++r) {
      int row = r + 8 * hh, col = t * 16 + n;
      Ax[row][col] = f2bf(c[r] * Xs[row][col]);
    }
  }
  {                                                     // hid = relu(ea @ W1k + b1k)
    v16u a = ldA(&Aea[0][0], 32, lane);
    v16u b = ldBp(W1kp, wave, lane);
    v8f c = {};
    c = wmma_bf16(a, b, c);
#pragma unroll
    for (int r = 0; r < 8; ++r) {
      int row = r + 8 * hh, col = wave * 16 + n;
      Hid[row][col] = f2bf(fmaxf(c[r] + b1k[col], 0.f));
    }
  }
  __syncthreads();
  {                                                     // rad = hid @ W2k
    v8f c = {};
#pragma unroll
    for (int kk = 0; kk < 2; ++kk) {
      v16u a = ldA(&Hid[0][kk * 32], HIDv, lane);
      v16u b = ldBp(W2kp, kk * 4 + wave, lane);
      c = wmma_bf16(a, b, c);
    }
#pragma unroll
    for (int r = 0; r < 8; ++r) Rad[r + 8 * hh][wave * 16 + n] = c[r];
  }
  {                                                     // k = (Ax @ Wlin_k) * rad
    v8f c = {};
#pragma unroll
    for (int kk = 0; kk < 4; ++kk) {
      v16u a = ldA(&Ax[0][kk * 32], DIN, lane);
      v16u b = ldBp(Wlinkp, kk * 4 + wave, lane);
      c = wmma_bf16(a, b, c);
    }
#pragma unroll
    for (int r = 0; r < 8; ++r) {
      int row = r + 8 * hh, col = wave * 16 + n;
      Kt[row][col] = c[r] * Rad[row][col];
    }
  }
  __syncthreads();
  {                                                     // alpha = qW[dst] . k (per head)
    int e = tid >> 3, h = tid & 7;
    const float* qp = qW + (long long)Dst[e] * DQv + h * 8;
    float s = 0.f;
#pragma unroll
    for (int i = 0; i < 8; ++i) s += qp[i] * Kt[e][h * 8 + i];
    alpha[(long long)(e0 + e) * NH + h] = s;
    atomicMaxF(&amax[Dst[e] * NH + h], s);
  }
}

// ---- pass B: exp + denom ---------------------------------------------------

__global__ void passB_kernel(const int* edge_index, const float* amax,
                             float* alpha, float* denom) {
  long long idx = (long long)blockIdx.x * blockDim.x + threadIdx.x;
  if (idx >= (long long)EE * NH) return;
  int e = (int)(idx >> 3), h = (int)(idx & 7);
  int d = edge_index[EE + e];
  float ex = __expf(alpha[idx] - amax[d * NH + h]);
  alpha[idx] = ex;
  atomicAdd(&denom[d * NH + h], ex);
}

// ---- pass C: recompute v, weighted atomic scatter --------------------------

__global__ __launch_bounds__(256) void passC_kernel(
    const float* node_attr, const float* edge_attr, const float* edge_sh,
    const float* b1v, const int* edge_index,
    const unsigned short* WangvTp, const unsigned short* W1vp,
    const unsigned short* W2vp, const unsigned short* Wlinvp,
    const float* alpha, const float* denom, float* acc) {
  __shared__ float          Xs[16][DIN];
  __shared__ unsigned short Ax[16][DIN];
  __shared__ unsigned short Ash[16][32];
  __shared__ unsigned short Aea[16][32];
  __shared__ unsigned short Hid[16][HIDv];
  __shared__ float          Rad[16][DVv];
  __shared__ float          Vt[16][DVv];
  __shared__ float          Attn[16][NH];
  __shared__ int Src[16], Dst[16];

  const int e0 = blockIdx.x * 16, tid = threadIdx.x;
  if (tid < 16) { Src[tid] = edge_index[e0 + tid]; Dst[tid] = edge_index[EE + e0 + tid]; }
  for (int idx = tid; idx < 16 * 32; idx += 256) {
    int r = idx >> 5, c = idx & 31;
    Aea[r][c] = (c < EAD) ? f2bf(edge_attr[(long long)(e0 + r) * EAD + c]) : (unsigned short)0;
    Ash[r][c] = (c < SHD) ? f2bf(edge_sh[(long long)(e0 + r) * SHD + c])  : (unsigned short)0;
  }
  __syncthreads();
  for (int idx = tid; idx < 16 * DIN; idx += 256) {
    int r = idx >> 7, c = idx & 127;
    Xs[r][c] = node_attr[(long long)Src[r] * DIN + c];
  }
  if (tid < 128) {
    int e = tid >> 3, h = tid & 7;
    Attn[e][h] = alpha[(long long)(e0 + e) * NH + h] / denom[Dst[e] * NH + h];
  }
  __syncthreads();

  const int wave = tid >> 5, lane = tid & 31;
  const int n = lane & 15, hh = (lane >> 4) & 1;

  {                                                     // g_v, fused with x -> Ax
    v16u a = ldA(&Ash[0][0], 32, lane);
    v16u b = ldBp(WangvTp, wave, lane);
    v8f c = {};
    c = wmma_bf16(a, b, c);
#pragma unroll
    for (int r = 0; r < 8; ++r) {
      int row = r + 8 * hh, col = wave * 16 + n;
      Ax[row][col] = f2bf(c[r] * Xs[row][col]);
    }
  }
  if (wave < 4) {                                       // hid
    v16u a = ldA(&Aea[0][0], 32, lane);
    v16u b = ldBp(W1vp, wave, lane);
    v8f c = {};
    c = wmma_bf16(a, b, c);
#pragma unroll
    for (int r = 0; r < 8; ++r) {
      int row = r + 8 * hh, col = wave * 16 + n;
      Hid[row][col] = f2bf(fmaxf(c[r] + b1v[col], 0.f));
    }
  }
  __syncthreads();
  {                                                     // rad_v
    v8f c = {};
#pragma unroll
    for (int kk = 0; kk < 2; ++kk) {
      v16u a = ldA(&Hid[0][kk * 32], HIDv, lane);
      v16u b = ldBp(W2vp, kk * 8 + wave, lane);
      c = wmma_bf16(a, b, c);
    }
#pragma unroll
    for (int r = 0; r < 8; ++r) Rad[r + 8 * hh][wave * 16 + n] = c[r];
  }
  {                                                     // v = (Ax @ Wlin_v) * rad
    v8f c = {};
#pragma unroll
    for (int kk = 0; kk < 4; ++kk) {
      v16u a = ldA(&Ax[0][kk * 32], DIN, lane);
      v16u b = ldBp(Wlinvp, kk * 8 + wave, lane);
      c = wmma_bf16(a, b, c);
    }
#pragma unroll
    for (int r = 0; r < 8; ++r) {
      int row = r + 8 * hh, col = wave * 16 + n;
      Vt[row][col] = c[r] * Rad[row][col];
    }
  }
  __syncthreads();
  for (int idx = tid; idx < 16 * DVv; idx += 256) {     // attn-weighted scatter
    int e = idx >> 7, c = idx & 127;
    atomicAdd(&acc[(long long)Dst[e] * DVv + c], Attn[e][c >> 4] * Vt[e][c]);
  }
}

// ---- pass D: out = acc @ Wout (N = 3125 * 16 exactly -> no guards) ---------

__global__ __launch_bounds__(256) void passD_kernel(const float* acc,
                                                    const unsigned short* Woutp,
                                                    float* out) {
  __shared__ unsigned short As[16][DIN];
  int m0 = blockIdx.x * 16, tid = threadIdx.x;
  for (int idx = tid; idx < 16 * DIN; idx += 256) {
    int r = idx >> 7, c = idx & 127;
    As[r][c] = f2bf(acc[(long long)(m0 + r) * DIN + c]);
  }
  __syncthreads();
  int wave = tid >> 5, lane = tid & 31;
  int n = lane & 15, hh = (lane >> 4) & 1;
  v8f c = {};
#pragma unroll
  for (int kk = 0; kk < 4; ++kk) {
    v16u a = ldA(&As[0][kk * 32], DIN, lane);
    v16u b = ldBp(Woutp, kk * 8 + wave, lane);
    c = wmma_bf16(a, b, c);
  }
  int gm0 = m0 + 8 * hh;
#pragma unroll
  for (int r = 0; r < 8; ++r)
    out[(long long)(gm0 + r) * DVv + wave * 16 + n] = c[r];
}

// ---- launcher --------------------------------------------------------------

extern "C" void kernel_launch(void* const* d_in, const int* in_sizes, int n_in,
                              void* d_out, int out_size, void* d_ws, size_t ws_size,
                              hipStream_t stream) {
  (void)in_sizes; (void)n_in; (void)out_size; (void)ws_size;
  const float* node_attr  = (const float*)d_in[0];
  const float* edge_attr  = (const float*)d_in[1];
  const float* edge_sh    = (const float*)d_in[2];
  const float* Wq         = (const float*)d_in[3];
  const float* Wang_k     = (const float*)d_in[4];
  const float* Wlin_k     = (const float*)d_in[5];
  const float* W1k        = (const float*)d_in[6];
  const float* b1k        = (const float*)d_in[7];
  const float* W2k        = (const float*)d_in[8];
  const float* Wang_v     = (const float*)d_in[9];
  const float* Wlin_v     = (const float*)d_in[10];
  const float* W1v        = (const float*)d_in[11];
  const float* b1v        = (const float*)d_in[12];
  const float* W2v        = (const float*)d_in[13];
  const float* Wdot       = (const float*)d_in[14];
  const float* Wout       = (const float*)d_in[15];
  const int*   edge_index = (const int*)d_in[16];

  char* ws = (char*)d_ws;
  size_t off = 0;
  auto alloc = [&](size_t bytes) -> void* {
    void* p = ws + off;
    off += (bytes + 255) & ~(size_t)255;
    return p;
  };
  float* qW    = (float*)alloc((size_t)NN * DQv * 4);
  float* alpha = (float*)alloc((size_t)EE * NH * 4);
  float* amax  = (float*)alloc((size_t)NN * NH * 4);
  float* denom = (float*)alloc((size_t)NN * NH * 4);
  float* acc   = (float*)alloc((size_t)NN * DVv * 4);
  float* Wq2f  = (float*)alloc(DIN * DQv * 4);
  // packed-B buffers: ktiles*ntiles*32 lanes*16 elems*2B
  unsigned short* Wq2p    = (unsigned short*)alloc(4 * 4 * 512 * 2);
  unsigned short* WangkTp = (unsigned short*)alloc(1 * 8 * 512 * 2);
  unsigned short* Wlinkp  = (unsigned short*)alloc(4 * 4 * 512 * 2);
  unsigned short* W1kp    = (unsigned short*)alloc(1 * 4 * 512 * 2);
  unsigned short* W2kp    = (unsigned short*)alloc(2 * 4 * 512 * 2);
  unsigned short* WangvTp = (unsigned short*)alloc(1 * 8 * 512 * 2);
  unsigned short* Wlinvp  = (unsigned short*)alloc(4 * 8 * 512 * 2);
  unsigned short* W1vp    = (unsigned short*)alloc(1 * 4 * 512 * 2);
  unsigned short* W2vp    = (unsigned short*)alloc(2 * 8 * 512 * 2);
  unsigned short* Woutp   = (unsigned short*)alloc(4 * 8 * 512 * 2);

  prep_kernel<<<1, 256, 0, stream>>>(Wq, Wdot, Wang_k, Wlin_k, W1k, W2k,
                                     Wang_v, Wlin_v, W1v, W2v, Wout, Wq2f,
                                     Wq2p, WangkTp, Wlinkp, W1kp, W2kp,
                                     WangvTp, Wlinvp, W1vp, W2vp, Woutp);
  init_kernel<<<(unsigned)((NN * DVv + 255) / 256), 256, 0, stream>>>(amax, denom, acc);
  qw_kernel<<<(unsigned)((NN + 31) / 32), 256, 0, stream>>>(node_attr, Wq2p, qW);
  passA_kernel<<<EE / 16, 128, 0, stream>>>(node_attr, edge_attr, edge_sh, b1k, edge_index,
                                            WangkTp, W1kp, W2kp, Wlinkp, qW, alpha, amax);
  passB_kernel<<<(unsigned)(((size_t)EE * NH + 255) / 256), 256, 0, stream>>>(edge_index, amax,
                                                                              alpha, denom);
  passC_kernel<<<EE / 16, 256, 0, stream>>>(node_attr, edge_attr, edge_sh, b1v, edge_index,
                                            WangvTp, W1vp, W2vp, Wlinvp, alpha, denom, acc);
  passD_kernel<<<3125, 256, 0, stream>>>(acc, Woutp, (float*)d_out);
}
